// GraphAttentionLayer_27633819583027
// MI455X (gfx1250) — matrix operand order repaired and checked by reading, not compiled, not run on hardware
//
#include <hip/hip_runtime.h>
#include <hip/hip_bf16.h>

// ---------- types ----------
typedef __attribute__((ext_vector_type(16))) __bf16 v16bf;
typedef __attribute__((ext_vector_type(8)))  float  v8f;
typedef __attribute__((ext_vector_type(8)))  int    v8i;
typedef __attribute__((ext_vector_type(4)))  int    v4i;
typedef __attribute__((ext_vector_type(4)))  float  v4f;

#define N_NODES 8192
#define IN_F    512
#define OUT_F   128
#define NUM_CLS 10

// softmax fixed shift: exp(s - 48) == exp2(s*log2e - 48*log2e)
#define L2E      1.44269504088896f
#define SHIFT_L2 69.2493619626703f   // 48 * log2(e)

// ---------- bf16 helpers (RNE) ----------
__device__ __forceinline__ unsigned f2bf_u(float x) {
    union { float f; unsigned u; } v; v.f = x;
    unsigned r = v.u + 0x7FFFu + ((v.u >> 16) & 1u);
    return r >> 16;
}
__device__ __forceinline__ float bf2f(unsigned b) {
    union { float f; unsigned u; } v; v.u = b << 16;
    return v.f;
}
__device__ __forceinline__ unsigned pack_bf2(float a, float b) {
    return f2bf_u(a) | (f2bf_u(b) << 16);
}

__device__ __forceinline__ v8f wmma_bf16(v8i a, v8i b, v8f c) {
    v16bf av = __builtin_bit_cast(v16bf, a);
    v16bf bv = __builtin_bit_cast(v16bf, b);
    return __builtin_amdgcn_wmma_f32_16x16x32_bf16(
        false, av, false, bv, (short)0, c, false, false);
}

// =====================================================================
// Kernel 1: Wh = h @ W  (fp32) via bf16 WMMA hi/lo split (3 WMMAs/step).
// =====================================================================
__global__ __launch_bounds__(256) void k_wh(const float* __restrict__ h,
                                            const float* __restrict__ W,
                                            float* __restrict__ Wh) {
    const int lane = threadIdx.x & 31;
    const int wave = threadIdx.x >> 5;
    const int i0   = blockIdx.x * 16;
    const int f0   = wave * 16;
    const int m    = lane & 15;
    const int half = lane >> 4;

    v8f acc = {};
    for (int k = 0; k < IN_F; k += 32) {
        const float* hrow = h + (size_t)(i0 + m) * IN_F + k + half * 8;
        float af[16];
#pragma unroll
        for (int t = 0; t < 8; ++t) af[t] = hrow[t];
#pragma unroll
        for (int t = 0; t < 8; ++t) af[8 + t] = hrow[16 + t];

        v8i ahi, alo;
#pragma unroll
        for (int t = 0; t < 8; ++t) {
            float x0 = af[2 * t], x1 = af[2 * t + 1];
            unsigned h0 = f2bf_u(x0), h1 = f2bf_u(x1);
            ahi[t] = (int)(h0 | (h1 << 16));
            alo[t] = (int)pack_bf2(x0 - bf2f(h0), x1 - bf2f(h1));
        }
        v8i bhi, blo;
#pragma unroll
        for (int v = 0; v < 8; ++v) {
            int kk = k + half * 8 + 2 * v + ((v & 4) << 1);
            float x0 = W[(size_t)kk * OUT_F + f0 + m];
            float x1 = W[(size_t)(kk + 1) * OUT_F + f0 + m];
            unsigned h0 = f2bf_u(x0), h1 = f2bf_u(x1);
            bhi[v] = (int)(h0 | (h1 << 16));
            blo[v] = (int)pack_bf2(x0 - bf2f(h0), x1 - bf2f(h1));
        }
        acc = wmma_bf16(ahi, bhi, acc);
        acc = wmma_bf16(ahi, blo, acc);
        acc = wmma_bf16(alo, bhi, acc);
    }
#pragma unroll
    for (int r = 0; r < 8; ++r)
        Wh[(size_t)(i0 + half * 8 + r) * OUT_F + f0 + m] = acc[r];
}

// =====================================================================
// Kernel 2: Wh1/Wh2 = Wh @ a halves
// =====================================================================
__global__ void k_av(const float* __restrict__ Wh, const float* __restrict__ a,
                     float* __restrict__ Wh1, float* __restrict__ Wh2) {
    int i = blockIdx.x * blockDim.x + threadIdx.x;
    if (i >= N_NODES) return;
    const float* r = Wh + (size_t)i * OUT_F;
    float s1 = 0.f, s2 = 0.f;
#pragma unroll 4
    for (int f = 0; f < OUT_F; ++f) {
        float v = r[f];
        s1 += v * a[f];
        s2 += v * a[OUT_F + f];
    }
    Wh1[i] = s1;
    Wh2[i] = s2;
}

// =====================================================================
// Kernel 3: pre-swizzle Wh into WMMA B-fragment order (bf16).
// WhB[((sub32*8 + tile)*32 + lane)*8 + v], sub32 = 32-key sub-chunk
// =====================================================================
__global__ void k_swz(const float* __restrict__ Wh, int* __restrict__ WhB) {
    int g = blockIdx.x * blockDim.x + threadIdx.x;   // 0..65535
    int lane = g & 31;
    int tile = (g >> 5) & 7;
    int ch   = g >> 8;                               // 0..255
    int n = lane & 15, half = lane >> 4;
    int f = tile * 16 + n;
    int k0 = ch * 32 + half * 8;
    int* outp = WhB + (size_t)g * 8;
#pragma unroll
    for (int v = 0; v < 8; ++v) {
        float x0 = Wh[(size_t)(k0 + 2 * v + ((v & 4) << 1)) * OUT_F + f];
        float x1 = Wh[(size_t)(k0 + 2 * v + ((v & 4) << 1) + 1) * OUT_F + f];
        outp[v] = (int)pack_bf2(x0, x1);
    }
}

// =====================================================================
// Kernel 4: fused masked-leakyrelu + fixed-shift softmax + P@Wh (WMMA)
//           + ELU + classifier.  16 query rows per workgroup, 8 waves
//           (wave w owns features [16w,16w+16)).  128 chunks of 64 keys.
// =====================================================================
__global__ __launch_bounds__(256) void k_attn(
    const int* __restrict__ dist, const int* __restrict__ WhB,
    const float* __restrict__ Wh1v, const float* __restrict__ Wh2v,
    const float* __restrict__ Wc, const float* __restrict__ bc,
    float* __restrict__ out) {
    __shared__ unsigned short Pb[2][16][64];   // bf16 P tiles (double buffer)
    __shared__ float Lb[16];                   // per-row softmax denominators
    __shared__ float Hu[16][OUT_F];            // h_update tile

    const int lane = threadIdx.x & 31;
    const int wave = threadIdx.x >> 5;
    const int i0   = blockIdx.x * 16;
    const int half = lane >> 4;
    const int col  = lane & 15;
    const int r_mine = wave * 2 + half;        // row this thread scores
    const int k_mine = col * 4;                // 4 keys per thread per chunk

    const float wh1 = Wh1v[i0 + r_mine];
    const int* distrow = dist + (size_t)(i0 + r_mine) * N_NODES;

    float l_acc = 0.f;                         // thread-local partial row sum
    v8f acc = {};

    for (int ch = 0; ch < 128; ++ch) {
        const int k0  = ch * 64;
        const int buf = ch & 1;
        if (ch < 126)
            __builtin_prefetch(distrow + k0 + 128 + k_mine, 0, 0);

        // ---- 4 masked scores -> p = exp2(s*log2e - 48*log2e) ----
        v4i d  = *(const v4i*)(distrow + k0 + k_mine);
        v4f w2 = *(const v4f*)(Wh2v + k0 + k_mine);
        float p[4];
#pragma unroll
        for (int j = 0; j < 4; ++j) {
            float x = wh1 + w2[j];
            float s = fmaxf(x, 0.2f * x);              // leaky_relu
            float t = fmaf(s, L2E, -SHIFT_L2);
            float arg = (d[j] > 0) ? t : -200.0f;      // mask -> exp2 == 0
            p[j] = __builtin_amdgcn_exp2f(arg);
        }
        l_acc += (p[0] + p[1]) + (p[2] + p[3]);

        unsigned long long pk =
            (unsigned long long)pack_bf2(p[0], p[1]) |
            ((unsigned long long)pack_bf2(p[2], p[3]) << 32);
        *(unsigned long long*)&Pb[buf][r_mine][k_mine] = pk;
        __syncthreads();   // single barrier/chunk; LDS double-buffered

        // ---- A fragments (16-bit A 16x32 layout), 2 sub-tiles ----
        const unsigned short* pr0 = &Pb[buf][col][half * 8];
        v4i a00 = *(const v4i*)pr0;
        v4i a01 = *(const v4i*)(pr0 + 16);
        const unsigned short* pr1 = &Pb[buf][col][32 + half * 8];
        v4i a10 = *(const v4i*)pr1;
        v4i a11 = *(const v4i*)(pr1 + 16);
        v8i A0, A1;
        A0[0]=a00.x; A0[1]=a00.y; A0[2]=a00.z; A0[3]=a00.w;
        A0[4]=a01.x; A0[5]=a01.y; A0[6]=a01.z; A0[7]=a01.w;
        A1[0]=a10.x; A1[1]=a10.y; A1[2]=a10.z; A1[3]=a10.w;
        A1[4]=a11.x; A1[5]=a11.y; A1[6]=a11.z; A1[7]=a11.w;

        // ---- B fragments: coalesced pre-swizzled ----
        const int* bp0 = WhB + ((((size_t)(2 * ch) * 8 + wave) * 32 + lane) * 8);
        const int* bp1 = bp0 + 8 * 32 * 8;     // next 32-key sub-chunk
        v4i b00 = *(const v4i*)bp0;
        v4i b01 = *(const v4i*)(bp0 + 4);
        v4i b10 = *(const v4i*)bp1;
        v4i b11 = *(const v4i*)(bp1 + 4);
        v8i B0, B1;
        B0[0]=b00.x; B0[1]=b00.y; B0[2]=b00.z; B0[3]=b00.w;
        B0[4]=b01.x; B0[5]=b01.y; B0[6]=b01.z; B0[7]=b01.w;
        B1[0]=b10.x; B1[1]=b10.y; B1[2]=b10.z; B1[3]=b10.w;
        B1[4]=b11.x; B1[5]=b11.y; B1[6]=b11.z; B1[7]=b11.w;

        acc = wmma_bf16(A0, B0, acc);
        acc = wmma_bf16(A1, B1, acc);
    }

    // ---- one-time row-sum reduction (16 lanes of half-wave) ----
#pragma unroll
    for (int off = 1; off < 16; off <<= 1)
        l_acc += __shfl_xor(l_acc, off, 32);
    if (col == 0) Lb[r_mine] = l_acc;
    __syncthreads();

    const float* lp = &Lb[half * 8];
#pragma unroll
    for (int r = 0; r < 8; ++r) {
        float x = acc[r] / lp[r];
        Hu[half * 8 + r][wave * 16 + col] = x > 0.f ? x : (__expf(x) - 1.f);
    }
    __syncthreads();

    // ---- fused classifier: y = Hu @ Wc^T + bc ----
    int t = threadIdx.x;
    if (t < 16 * NUM_CLS) {
        int row = t / NUM_CLS, cls = t % NUM_CLS;
        const float* hr = &Hu[row][0];
        const float* wr = Wc + (size_t)cls * OUT_F;
        float s = bc[cls];
#pragma unroll 4
        for (int f = 0; f < OUT_F; ++f) s += hr[f] * wr[f];
        out[(size_t)(i0 + row) * NUM_CLS + cls] = s;
    }
}

// =====================================================================
extern "C" void kernel_launch(void* const* d_in, const int* in_sizes, int n_in,
                              void* d_out, int out_size, void* d_ws, size_t ws_size,
                              hipStream_t stream) {
    const float* h    = (const float*)d_in[0];
    const int*   dist = (const int*)d_in[1];
    const float* W    = (const float*)d_in[2];
    const float* a    = (const float*)d_in[3];
    const float* Wc   = (const float*)d_in[4];
    const float* bc   = (const float*)d_in[5];
    float* out = (float*)d_out;

    char* ws = (char*)d_ws;
    float* Wh  = (float*)ws;                               // 4 MB
    int*   WhB = (int*)(ws + (size_t)(4 << 20));           // 2 MB
    float* Wh1 = (float*)(ws + (size_t)(6 << 20));         // 32 KB
    float* Wh2 = (float*)(ws + (size_t)(6 << 20) + 32768); // 32 KB

    k_wh  <<<N_NODES / 16, 256, 0, stream>>>(h, W, Wh);
    k_av  <<<N_NODES / 256, 256, 0, stream>>>(Wh, a, Wh1, Wh2);
    k_swz <<<256, 256, 0, stream>>>(Wh, WhB);
    k_attn<<<N_NODES / 16, 256, 0, stream>>>(dist, WhB, Wh1, Wh2, Wc, bc, out);
}
